// FastWeightAttention_27530740368033
// MI455X (gfx1250) — matrix operand order repaired and checked by reading, not compiled, not run on hardware
//
#include <hip/hip_runtime.h>

// ---------------------------------------------------------------------------
// Problem constants (match reference: B=4, N=1024, D=1024, H=8, DH=64)
// ---------------------------------------------------------------------------
constexpr int kB = 4;
constexpr int kN = 1024;
constexpr int kD = 1024;
constexpr int kH = 8;
constexpr int kE = 64;   // DH

#define NS_A 3.4445f
#define NS_B (-4.775f)
#define NS_C 2.0315f
#define MAX_LR 0.01f

typedef __bf16 bf16_t;
typedef __attribute__((ext_vector_type(16))) __bf16 v16bf;
typedef __attribute__((ext_vector_type(8)))  float  v8f;

#define DEV __device__ __forceinline__

DEV float bf2f(bf16_t b) {
  unsigned short s = __builtin_bit_cast(unsigned short, b);
  unsigned int u = ((unsigned int)s) << 16;
  return __builtin_bit_cast(float, u);
}
DEV bf16_t f2bf(float f) {
  unsigned int u = __builtin_bit_cast(unsigned int, f);
  unsigned int r = (u + 0x7FFFu + ((u >> 16) & 1u)) >> 16;
  return __builtin_bit_cast(bf16_t, (unsigned short)r);
}
DEV float sigmoidf_(float x) { return 1.f / (1.f + __expf(-x)); }

// ---------------------------------------------------------------------------
// Batched BF16 GEMM with WMMA:  C = alpha * op(A) * op(B)
//   OPA/OPB template: 0 = stored row-major (M,K)/(K,N); 1 = stored transposed.
//   TMv x TNv tile, K-step 32, 256 threads = 8 waves, wave does 32x64
//   (8 x v_wmma_f32_16x16x32_bf16 per K-step).
// Software-pipelined double-buffered LDS: next tile's global_load_b128s are
// issued before the current tile's WMMAs, regs are committed to the alternate
// LDS buffer after compute, one barrier per K-step.
// Register-pressure discipline (the backend caps this kernel near 128 VGPRs
// and spills beyond it): (a) only ONE A-fragment is held at a time (B-frags
// re-read from LDS: 20 vs 12 ds_load_b128 per 8 WMMAs — cheaper than scratch
// spills); (b) all tile offsets are 32-bit ints (per-batch extents <= 1M
// elements) so global loads use scalar-base + 32-bit voffset addressing.
// All M,N,K are multiples of 64 and operands 16B-aligned -> staging unchecked.
// ---------------------------------------------------------------------------
struct GemmP {
  const bf16_t* A;
  const bf16_t* B;
  float*  Cf;     // optional f32 output
  bf16_t* Cb;     // optional bf16 output
  int M, N, K, lda, ldb, ldc;
  int bdiv;
  long long sAb, sAh, sBb, sBh, sCb, sCh;
  float alpha;
};

#define TK 32
#define LST 40   // LDS row stride in bf16 elements (80B: 16B aligned, padded)

template <int OPA, int OPB, int TMv, int TNv>
__global__ __launch_bounds__(256)
__attribute__((amdgpu_waves_per_eu(1))) void gemm_bf16_wmma(GemmP p) {
  constexpr int NCA = TMv / 64;   // 16B chunks per thread for A tile
  constexpr int NCB = TNv / 64;   // 16B chunks per thread for B tile
  __shared__ __align__(16) bf16_t As[2][TMv * LST];
  __shared__ __align__(16) bf16_t Bs[2][TNv * LST];

  constexpr int WGX = (TNv == 128) ? 2 : 1;  // wave grid in cols

  const int tid  = threadIdx.x;
  const int wave = tid >> 5;
  const int lane = tid & 31;
  const int half = lane >> 4;   // 0: lanes 0-15, 1: lanes 16-31
  const int l16  = lane & 15;
  const int rowb = (wave / WGX) * 32;
  const int colb = (wave % WGX) * 64;

  const int z  = blockIdx.z;
  const int zb = z / p.bdiv, zh = z % p.bdiv;
  const bf16_t* Ap = p.A + (long long)zb * p.sAb + (long long)zh * p.sAh;
  const bf16_t* Bp = p.B + (long long)zb * p.sBb + (long long)zh * p.sBh;
  const long long coff = (long long)zb * p.sCb + (long long)zh * p.sCh;

  const int row0 = blockIdx.y * TMv;
  const int col0 = blockIdx.x * TNv;

  uint4 ra[NCA], rb[NCB];

  auto loadA = [&](int k0) {
#pragma unroll
    for (int i = 0; i < NCA; ++i) {
      int c = tid + i * 256;
      if constexpr (OPA == 0) {
        int m = c >> 2, kc = (c & 3) * 8;
        ra[i] = *(const uint4*)(Ap + ((row0 + m) * p.lda + (k0 + kc)));
      } else {                               // stored (K,M): contig in m
        int k = c / (TMv / 8), mc = (c % (TMv / 8)) * 8;
        ra[i] = *(const uint4*)(Ap + ((k0 + k) * p.lda + (row0 + mc)));
      }
    }
  };
  auto storeA = [&](int buf) {
#pragma unroll
    for (int i = 0; i < NCA; ++i) {
      int c = tid + i * 256;
      if constexpr (OPA == 0) {
        int m = c >> 2, kc = (c & 3) * 8;
        *(uint4*)(&As[buf][m * LST + kc]) = ra[i];
      } else {
        int k = c / (TMv / 8), mc = (c % (TMv / 8)) * 8;
        const bf16_t* pv = (const bf16_t*)&ra[i];
#pragma unroll
        for (int j = 0; j < 8; ++j) As[buf][(mc + j) * LST + k] = pv[j];
      }
    }
  };
  auto loadB = [&](int k0) {
#pragma unroll
    for (int i = 0; i < NCB; ++i) {
      int c = tid + i * 256;
      if constexpr (OPB == 0) {               // stored (K,N): contig in n
        int k = c / (TNv / 8), nc = (c % (TNv / 8)) * 8;
        rb[i] = *(const uint4*)(Bp + ((k0 + k) * p.ldb + (col0 + nc)));
      } else {                                // stored (N,K): contig in k
        int n = c >> 2, kc = (c & 3) * 8;
        rb[i] = *(const uint4*)(Bp + ((col0 + n) * p.ldb + (k0 + kc)));
      }
    }
  };
  auto storeB = [&](int buf) {
#pragma unroll
    for (int i = 0; i < NCB; ++i) {
      int c = tid + i * 256;
      if constexpr (OPB == 0) {
        int k = c / (TNv / 8), nc = (c % (TNv / 8)) * 8;
        const bf16_t* pv = (const bf16_t*)&rb[i];
#pragma unroll
        for (int j = 0; j < 8; ++j) Bs[buf][(nc + j) * LST + k] = pv[j];
      } else {
        int n = c >> 2, kc = (c & 3) * 8;
        *(uint4*)(&Bs[buf][n * LST + kc]) = rb[i];
      }
    }
  };

  v8f acc[2][4];
  const v8f vz = {0.f, 0.f, 0.f, 0.f, 0.f, 0.f, 0.f, 0.f};
#pragma unroll
  for (int mt = 0; mt < 2; ++mt)
#pragma unroll
    for (int nt = 0; nt < 4; ++nt) acc[mt][nt] = vz;

  // prologue: stage tile 0
  loadA(0);
  loadB(0);
  storeA(0);
  storeB(0);
  __syncthreads();

  const int nsteps = p.K / TK;
  for (int s = 0; s < nsteps; ++s) {
    const int cur = s & 1;
    const bool more = (s + 1) < nsteps;
    if (more) {                      // issue next tile's global loads early
      loadA((s + 1) * TK);
      loadB((s + 1) * TK);
    }

    // One A-fragment live at a time (pressure). A frag (16x32 bf16, ISA
    // layout): element e -> K = (e%8)+(e/8)*16 + half*8 => two 16B chunks.
#pragma unroll
    for (int mt = 0; mt < 2; ++mt) {
      union { v16bf v; uint4 q[2]; } fa;
      const bf16_t* ap = &As[cur][(rowb + mt * 16 + l16) * LST + half * 8];
      fa.q[0] = *(const uint4*)ap;
      fa.q[1] = *(const uint4*)(ap + 16);
#pragma unroll
      for (int nt = 0; nt < 4; ++nt) {
        // B fragment (32x16 bf16): element e -> K = (e%8)+(e/8)*16 + half*8.
        union { v16bf v; uint4 q[2]; } fb;
        const bf16_t* bp = &Bs[cur][(colb + nt * 16 + l16) * LST + half * 8];
        fb.q[0] = *(const uint4*)bp;
        fb.q[1] = *(const uint4*)(bp + 16);
        acc[mt][nt] = __builtin_amdgcn_wmma_f32_16x16x32_bf16(
            false, fa.v, false, fb.v, (short)0, acc[mt][nt], false, false);
      }
    }

    if (more) {                      // commit next tile to the other buffer
      storeA(cur ^ 1);
      storeB(cur ^ 1);
    }
    __syncthreads();
  }

  // ---- store: C/D layout — lane col = l16, VGPR r -> row = r + 8*half ----
#pragma unroll
  for (int mt = 0; mt < 2; ++mt)
#pragma unroll
    for (int nt = 0; nt < 4; ++nt)
#pragma unroll
      for (int r = 0; r < 8; ++r) {
        int gm = row0 + rowb + mt * 16 + half * 8 + r;
        int gn = col0 + colb + nt * 16 + l16;
        if (gm < p.M && gn < p.N) {
          float v = p.alpha * acc[mt][nt][r];
          long long ci = coff + (gm * p.ldc + gn);
          if (p.Cf) p.Cf[ci] = v;
          if (p.Cb) p.Cb[ci] = f2bf(v);
        }
      }
}

// ---------------------------------------------------------------------------
// f32 -> bf16 convert
// ---------------------------------------------------------------------------
__global__ void cvt_bf16(const float* in, bf16_t* out, long long n) {
  long long i = (long long)blockIdx.x * blockDim.x + threadIdx.x;
  if (i < n) out[i] = f2bf(in[i]);
}

// ---------------------------------------------------------------------------
// RMSNorm + gates + learning rates.
// One block per (b,n) row. Writes t_bf, tqk_bf = t*lr, tv_bf = t*mlr
// (rows zeroed at n==N-1 to implement the [:-1] slicing), gates, lr, mlr.
// ---------------------------------------------------------------------------
__global__ __launch_bounds__(256) void rmsnorm_gates(
    const float* tokens, const float* rms_w, const float* w_gates,
    const float* w_lr, bf16_t* t_bf, bf16_t* tqk_bf, bf16_t* tv_bf,
    float* gates, float* lrv, float* mlrv) {
  const int row = blockIdx.x;        // b*N + n
  const int b = row / kN, n = row % kN;
  const float* tok = tokens + (long long)row * kD;

  __shared__ float trow[kD];
  __shared__ float red[256];
  __shared__ float outs[10];
  const int tid = threadIdx.x;

  float ss = 0.f;
  for (int d = tid; d < kD; d += 256) { float x = tok[d]; ss += x * x; }
  red[tid] = ss;
  __syncthreads();
  for (int s = 128; s > 0; s >>= 1) {
    if (tid < s) red[tid] += red[tid + s];
    __syncthreads();
  }
  const float rs = rsqrtf(red[0] / (float)kD + 1.1920929e-7f);
  __syncthreads();

  for (int d = tid; d < kD; d += 256) {
    float tv_ = tok[d] * rs * rms_w[d];
    trow[d] = tv_;
    t_bf[(long long)row * kD + d] = f2bf(tv_);
  }
  __syncthreads();

  float acc[10];
#pragma unroll
  for (int o = 0; o < 10; ++o) acc[o] = 0.f;
  for (int d = tid; d < kD; d += 256) {
    float tv_ = trow[d];
#pragma unroll
    for (int h = 0; h < kH; ++h) acc[h] += tv_ * w_gates[d * kH + h];
    acc[8] += tv_ * w_lr[d * 2 + 0];
    acc[9] += tv_ * w_lr[d * 2 + 1];
  }
  for (int o = 0; o < 10; ++o) {
    red[tid] = acc[o];
    __syncthreads();
    for (int s = 128; s > 0; s >>= 1) {
      if (tid < s) red[tid] += red[tid + s];
      __syncthreads();
    }
    if (tid == 0) outs[o] = red[0];
    __syncthreads();
  }

  const float lr  = sigmoidf_(outs[8]) * MAX_LR;
  const float mlr = sigmoidf_(outs[9]) * MAX_LR;
  if (tid < kH) gates[((long long)(b * kH + tid)) * kN + n] = sigmoidf_(outs[tid]);
  if (tid == 0) { lrv[row] = lr; mlrv[row] = mlr; }

  const bool last = (n == kN - 1);
  for (int d = tid; d < kD; d += 256) {
    float tv_ = last ? 0.f : trow[d];
    tqk_bf[(long long)row * kD + d] = f2bf(tv_ * lr);
    tv_bf[(long long)row * kD + d]  = f2bf(tv_ * mlr);
  }
}

// ---------------------------------------------------------------------------
// Causal softmax: one block per (b,h,i) row of score. attn written as bf16,
// zeros above the diagonal.
// ---------------------------------------------------------------------------
__global__ __launch_bounds__(128) void softmax_causal(const float* score,
                                                      bf16_t* attn) {
  const long long row = blockIdx.x;          // (b*H+h)*N + i
  const int i = (int)(row % kN);
  const float* s = score + row * kN;
  bf16_t* a = attn + row * kN;
  __shared__ float red[128];
  const int tid = threadIdx.x;

  float m = -3.4e38f;
  for (int j = tid; j <= i; j += 128) m = fmaxf(m, s[j]);
  red[tid] = m;
  __syncthreads();
  for (int st = 64; st > 0; st >>= 1) {
    if (tid < st) red[tid] = fmaxf(red[tid], red[tid + st]);
    __syncthreads();
  }
  m = red[0];
  __syncthreads();

  float sum = 0.f;
  for (int j = tid; j <= i; j += 128) sum += __expf(s[j] - m);
  red[tid] = sum;
  __syncthreads();
  for (int st = 64; st > 0; st >>= 1) {
    if (tid < st) red[tid] += red[tid + st];
    __syncthreads();
  }
  const float inv = 1.f / red[0];

  for (int j = tid; j < kN; j += 128)
    a[j] = (j <= i) ? f2bf(__expf(s[j] - m) * inv) : f2bf(0.f);
}

// ---------------------------------------------------------------------------
// out = attn@v already in out_raw (f32). Apply gates; emit o layouts:
//  o_pred (B,N,H,E) bf16 for pred GEMM; o_wo = o*mlr (B,H,N,E) bf16 for dwo.
// ---------------------------------------------------------------------------
__global__ void gate_out(const float* out_raw, const float* gates,
                         const float* mlrv, bf16_t* o_pred, bf16_t* o_wo) {
  long long idx = (long long)blockIdx.x * blockDim.x + threadIdx.x;
  const long long total = (long long)kB * kH * kN * kE;
  if (idx >= total) return;
  int e = (int)(idx & (kE - 1));
  long long t0 = idx >> 6;
  int n = (int)(t0 % kN);
  long long t1 = t0 / kN;
  int h = (int)(t1 % kH);
  int b = (int)(t1 / kH);
  float g = gates[((long long)(b * kH + h)) * kN + n];
  float o = out_raw[idx] * g;
  o_pred[(((long long)b * kN + n) * kH + h) * kE + e] = f2bf(o);
  o_wo[idx] = f2bf(o * mlrv[(long long)b * kN + n]);
}

// ---------------------------------------------------------------------------
// LayerNorm(tv) - pred -> error (bf16). One block per (b,n); row N-1 zeroed.
// tv row is tw[b, n+1, :].
// ---------------------------------------------------------------------------
__global__ __launch_bounds__(256) void ln_error(const float* tw,
                                                const float* pred,
                                                bf16_t* err) {
  const int row = blockIdx.x;   // b*N + n
  const int n = row % kN, b = row / kN;
  const int tid = threadIdx.x;
  if (n == kN - 1) {
    for (int d = tid; d < kD; d += 256) err[(long long)row * kD + d] = f2bf(0.f);
    return;
  }
  const float* x = tw + (long long)(b * kN + n + 1) * kD;
  __shared__ float r1[256], r2[256];
  float s1 = 0.f, s2 = 0.f;
  for (int d = tid; d < kD; d += 256) { float v = x[d]; s1 += v; s2 += v * v; }
  r1[tid] = s1; r2[tid] = s2;
  __syncthreads();
  for (int s = 128; s > 0; s >>= 1) {
    if (tid < s) { r1[tid] += r1[tid + s]; r2[tid] += r2[tid + s]; }
    __syncthreads();
  }
  const float mean = r1[0] / (float)kD;
  const float var = r2[0] / (float)kD - mean * mean;
  const float rs = rsqrtf(var + 1e-5f);
  for (int d = tid; d < kD; d += 256)
    err[(long long)row * kD + d] =
        f2bf((x[d] - mean) * rs - pred[(long long)row * kD + d]);
}

// ---------------------------------------------------------------------------
// du = dout * g (bf16) ; delta = sum_e dout * (out_raw*g). One thread/row.
// (dout row N-1 is exactly zero because error row N-1 is zero.)
// ---------------------------------------------------------------------------
__global__ void du_delta(const float* dout, const float* out_raw,
                         const float* gates, bf16_t* du, float* delta) {
  long long r = (long long)blockIdx.x * blockDim.x + threadIdx.x;
  const long long rows = (long long)kB * kH * kN;
  if (r >= rows) return;
  const float g = gates[r];
  const float* dp = dout + r * kE;
  const float* op = out_raw + r * kE;
  float dl = 0.f;
#pragma unroll 8
  for (int e = 0; e < kE; ++e) {
    float d = dp[e];
    du[r * kE + e] = f2bf(d * g);
    dl += d * op[e] * g;
  }
  delta[r] = dl;
}

// ---------------------------------------------------------------------------
// dscore = scale * a * (dattn - delta), written in place over attn (bf16).
// Zero pattern of a / delta makes the N-1 row & column exactly zero.
// ---------------------------------------------------------------------------
__global__ void dscore_k(bf16_t* attn, const float* dattn, const float* delta) {
  long long idx = (long long)blockIdx.x * blockDim.x + threadIdx.x;
  const long long total = (long long)kB * kH * kN * kN;
  if (idx >= total) return;
  float a = bf2f(attn[idx]);
  float v = 0.125f * a * (dattn[idx] - delta[idx / kN]);
  attn[idx] = f2bf(v);
}

// ---------------------------------------------------------------------------
// Newton-Schulz (5 iters) on 64x1024 matrices, one workgroup per matrix.
// Matrices 0..31: dwv (stored (1024,64), transposed in/out); 32..63: dwo.
// X kept in global scratch (ping-pong), A/Bm in LDS, all f32.
// ---------------------------------------------------------------------------
__global__ __launch_bounds__(256) void ns5_kernel(float* dwv, float* dwo,
                                                  float* X0g, float* X1g) {
  const int mat = blockIdx.x;      // 0..63
  const bool isV = mat < 32;
  const int bh = isV ? mat : mat - 32;
  float* G = isV ? (dwv + (long long)bh * kD * kE)
                 : (dwo + (long long)bh * kE * kD);
  float* Xa = X0g + (long long)mat * kE * kN;
  float* Xb = X1g + (long long)mat * kE * kN;

  __shared__ float Alds[64 * 64];
  __shared__ float Blds[64 * 64];
  __shared__ float red[256];
  const int tid = threadIdx.x;
  const int total = kE * kN;       // 65536

  // load (transpose dwv) + frobenius norm
  float ss = 0.f;
  for (int idx = tid; idx < total; idx += 256) {
    int e = idx >> 10, n = idx & (kN - 1);
    float g = isV ? G[(long long)n * kE + e] : G[idx];
    Xa[idx] = g;
    ss += g * g;
  }
  red[tid] = ss;
  __syncthreads();
  for (int s = 128; s > 0; s >>= 1) {
    if (tid < s) red[tid] += red[tid + s];
    __syncthreads();
  }
  const float inv = 1.f / (sqrtf(red[0]) + 1e-7f);
  __syncthreads();
  for (int idx = tid; idx < total; idx += 256) Xa[idx] *= inv;
  __threadfence();
  __syncthreads();

  float* Xc = Xa;
  float* Xn = Xb;
  for (int it = 0; it < 5; ++it) {
    // A = X X^T (64x64)
    for (int idx = tid; idx < 4096; idx += 256) {
      int i = idx >> 6, j = idx & 63;
      const float* ri = Xc + i * kN;
      const float* rj = Xc + j * kN;
      float a = 0.f;
      for (int k2 = 0; k2 < kN; ++k2) a += ri[k2] * rj[k2];
      Alds[idx] = a;
    }
    __syncthreads();
    // Bm = NS_B*A + NS_C*A@A
    for (int idx = tid; idx < 4096; idx += 256) {
      int i = idx >> 6, j = idx & 63;
      float a2 = 0.f;
#pragma unroll 8
      for (int kk = 0; kk < 64; ++kk) a2 += Alds[i * 64 + kk] * Alds[kk * 64 + j];
      Blds[idx] = NS_B * Alds[idx] + NS_C * a2;
    }
    __syncthreads();
    // X = NS_A*X + Bm@X
    for (int idx = tid; idx < total; idx += 256) {
      int i = idx >> 10, c = idx & (kN - 1);
      float a = NS_A * Xc[idx];
#pragma unroll 8
      for (int kk = 0; kk < 64; ++kk) a += Blds[i * 64 + kk] * Xc[kk * kN + c];
      Xn[idx] = a;
    }
    __threadfence();
    __syncthreads();
    float* t = Xc; Xc = Xn; Xn = t;
  }

  for (int idx = tid; idx < total; idx += 256) {
    int e = idx >> 10, n = idx & (kN - 1);
    if (isV) G[(long long)n * kE + e] = Xc[idx];
    else     G[idx] = Xc[idx];
  }
}

// ---------------------------------------------------------------------------
// Host launcher
// ---------------------------------------------------------------------------
template <int OPA, int OPB, int TMv, int TNv>
static void launch_gemm(hipStream_t s, const bf16_t* A, const bf16_t* B,
                        float* Cf, bf16_t* Cb, int M, int N, int K, int lda,
                        int ldb, int ldc, int bdiv, long long sAb,
                        long long sAh, long long sBb, long long sBh,
                        long long sCb, long long sCh, int batches,
                        float alpha) {
  GemmP p{A, B, Cf, Cb, M, N, K, lda, ldb, ldc, bdiv,
          sAb, sAh, sBb, sBh, sCb, sCh, alpha};
  dim3 g((N + TNv - 1) / TNv, (M + TMv - 1) / TMv, batches);
  gemm_bf16_wmma<OPA, OPB, TMv, TNv><<<g, dim3(256), 0, s>>>(p);
}

extern "C" void kernel_launch(void* const* d_in, const int* in_sizes, int n_in,
                              void* d_out, int out_size, void* d_ws,
                              size_t ws_size, hipStream_t stream) {
  (void)in_sizes; (void)n_in; (void)out_size; (void)ws_size;
  const float* tokens  = (const float*)d_in[0];
  const float* wq      = (const float*)d_in[1];
  const float* wk      = (const float*)d_in[2];
  const float* wv      = (const float*)d_in[3];
  const float* wo      = (const float*)d_in[4];
  const float* w_lr    = (const float*)d_in[5];
  const float* w_tgt   = (const float*)d_in[6];
  const float* w_gates = (const float*)d_in[7];
  const float* rms_w   = (const float*)d_in[8];

  float* pred   = (float*)d_out;                         // (B,N,D)
  float* dwq_o  = pred  + (long long)kB * kN * kD;       // (B,H,D,E)
  float* dwk_o  = dwq_o + (long long)kB * kH * kD * kE;
  float* dwv_o  = dwk_o + (long long)kB * kH * kD * kE;
  float* dwo_o  = dwv_o + (long long)kB * kH * kD * kE;  // (B,H,E,D)

  // ---- workspace layout (~345 MB) ----
  size_t off = 0;
  auto alloc = [&](size_t bytes) -> char* {
    char* p = (char*)d_ws + off;
    off += (bytes + 255) & ~(size_t)255;
    return p;
  };
  const long long BND = (long long)kB * kN * kD;
  const long long BHN = (long long)kB * kH * kN;
  const long long BHNE = BHN * kE;
  const long long BHNN = BHN * kN;

  bf16_t* t_bf   = (bf16_t*)alloc(BND * 2);
  bf16_t* tqk_bf = (bf16_t*)alloc(BND * 2);
  bf16_t* tv_bf  = (bf16_t*)alloc(BND * 2);
  bf16_t* wq_bf  = (bf16_t*)alloc((long long)kH * kD * kE * 2);
  bf16_t* wk_bf  = (bf16_t*)alloc((long long)kH * kD * kE * 2);
  bf16_t* wv_bf  = (bf16_t*)alloc((long long)kH * kD * kE * 2);
  bf16_t* wo_bf  = (bf16_t*)alloc((long long)kH * kE * kD * 2);
  bf16_t* wt_bf  = (bf16_t*)alloc((long long)kD * kD * 2);
  float*  gates  = (float*)alloc(BHN * 4);
  float*  lrv    = (float*)alloc((long long)kB * kN * 4);
  float*  mlrv   = (float*)alloc((long long)kB * kN * 4);
  bf16_t* q_bf   = (bf16_t*)alloc(BHNE * 2);
  bf16_t* k_bf   = (bf16_t*)alloc(BHNE * 2);
  bf16_t* v_bf   = (bf16_t*)alloc(BHNE * 2);
  float*  score  = (float*)alloc(BHNN * 4);   // reused for dattn
  bf16_t* attn   = (bf16_t*)alloc(BHNN * 2);  // reused for dscore (in place)
  float*  outr   = (float*)alloc(BHNE * 4);
  bf16_t* o_pred = (bf16_t*)alloc(BHNE * 2);
  bf16_t* o_wo   = (bf16_t*)alloc(BHNE * 2);
  float*  tw     = (float*)alloc(BND * 4);
  bf16_t* err_bf = (bf16_t*)alloc(BND * 2);
  float*  doutf  = (float*)alloc(BHNE * 4);
  bf16_t* du_bf  = (bf16_t*)alloc(BHNE * 2);
  bf16_t* dv_bf  = (bf16_t*)alloc(BHNE * 2);
  bf16_t* dq_bf  = (bf16_t*)alloc(BHNE * 2);
  bf16_t* dk_bf  = (bf16_t*)alloc(BHNE * 2);
  float*  delta  = (float*)alloc(BHN * 4);
  float*  X0g    = (float*)alloc((long long)64 * kE * kN * 4);
  float*  X1g    = (float*)alloc((long long)64 * kE * kN * 4);

  const long long WSZ = (long long)kH * kD * kE;
  auto cvt = [&](const float* in, bf16_t* out, long long n) {
    cvt_bf16<<<dim3((unsigned)((n + 255) / 256)), 256, 0, stream>>>(in, out, n);
  };
  cvt(wq, wq_bf, WSZ);
  cvt(wk, wk_bf, WSZ);
  cvt(wv, wv_bf, WSZ);
  cvt(wo, wo_bf, WSZ);
  cvt(w_tgt, wt_bf, (long long)kD * kD);

  // RMSNorm + gates + lr
  rmsnorm_gates<<<dim3(kB * kN), 256, 0, stream>>>(
      tokens, rms_w, w_gates, w_lr, t_bf, tqk_bf, tv_bf, gates, lrv, mlrv);

  const int BH = kB * kH;
  const long long ND = (long long)kN * kD, NE = (long long)kN * kE;
  const long long NN2 = (long long)kN * kN, DE = (long long)kD * kE;

  // q,k,v = t @ w*   (batched over b,h)  [narrow tiles: N=64]
  launch_gemm<0, 0, 256, 64>(stream, t_bf, wq_bf, nullptr, q_bf, kN, kE, kD,
                             kD, kE, kE, kH, ND, 0, 0, DE,
                             (long long)kH * NE, NE, BH, 1.f);
  launch_gemm<0, 0, 256, 64>(stream, t_bf, wk_bf, nullptr, k_bf, kN, kE, kD,
                             kD, kE, kE, kH, ND, 0, 0, DE,
                             (long long)kH * NE, NE, BH, 1.f);
  launch_gemm<0, 0, 256, 64>(stream, t_bf, wv_bf, nullptr, v_bf, kN, kE, kD,
                             kD, kE, kE, kH, ND, 0, 0, DE,
                             (long long)kH * NE, NE, BH, 1.f);

  // score = scale * q @ k^T   [wide tiles]
  launch_gemm<0, 1, 128, 128>(stream, q_bf, k_bf, score, nullptr, kN, kN, kE,
                              kE, kE, kN, kH, (long long)kH * NE, NE,
                              (long long)kH * NE, NE, (long long)kH * NN2, NN2,
                              BH, 0.125f);

  // attn = causal softmax(score)  (bf16)
  softmax_causal<<<dim3(kB * kH * kN), 128, 0, stream>>>(score, attn);

  // out_raw = attn @ v
  launch_gemm<0, 0, 256, 64>(stream, attn, v_bf, outr, nullptr, kN, kE, kN,
                             kN, kE, kE, kH, (long long)kH * NN2, NN2,
                             (long long)kH * NE, NE, (long long)kH * NE, NE,
                             BH, 1.f);

  // gate + layouts
  gate_out<<<dim3((unsigned)((BHNE + 255) / 256)), 256, 0, stream>>>(
      outr, gates, mlrv, o_pred, o_wo);

  // pred = o_pred(B,N,H*E) @ wo(H*E, D)   (batched over b)
  launch_gemm<0, 0, 128, 128>(stream, o_pred, wo_bf, pred, nullptr, kN, kD,
                              kH * kE, kH * kE, kD, kD, 1,
                              (long long)kN * kH * kE, 0, 0, 0, ND, 0, kB,
                              1.f);

  // tw = t @ w_target
  launch_gemm<0, 0, 128, 128>(stream, t_bf, wt_bf, tw, nullptr, kN, kD, kD,
                              kD, kD, kD, 1, ND, 0, 0, 0, ND, 0, kB, 1.f);

  // error = LN(tw shifted) - pred  (row N-1 zeroed)
  ln_error<<<dim3(kB * kN), 256, 0, stream>>>(tw, pred, err_bf);

  // dout = error @ wo^T  (per b,h)
  launch_gemm<0, 1, 256, 64>(stream, err_bf, wo_bf, doutf, nullptr, kN, kE,
                             kD, kD, kD, kE, kH, ND, 0, 0,
                             (long long)kE * kD, (long long)kH * NE, NE, BH,
                             1.f);

  // du, delta
  du_delta<<<dim3((unsigned)((BHN + 255) / 256)), 256, 0, stream>>>(
      doutf, outr, gates, du_bf, delta);

  // dv = a^T @ du
  launch_gemm<1, 0, 256, 64>(stream, attn, du_bf, nullptr, dv_bf, kN, kE, kN,
                             kN, kE, kE, kH, (long long)kH * NN2, NN2,
                             (long long)kH * NE, NE, (long long)kH * NE, NE,
                             BH, 1.f);

  // dattn = du @ v^T  (into score buffer)
  launch_gemm<0, 1, 128, 128>(stream, du_bf, v_bf, score, nullptr, kN, kN, kE,
                              kE, kE, kN, kH, (long long)kH * NE, NE,
                              (long long)kH * NE, NE, (long long)kH * NN2, NN2,
                              BH, 1.f);

  // dscore in place over attn
  dscore_k<<<dim3((unsigned)((BHNN + 255) / 256)), 256, 0, stream>>>(
      attn, score, delta);

  // dq = dscore @ k ; dk = dscore^T @ q
  launch_gemm<0, 0, 256, 64>(stream, attn, k_bf, nullptr, dq_bf, kN, kE, kN,
                             kN, kE, kE, kH, (long long)kH * NN2, NN2,
                             (long long)kH * NE, NE, (long long)kH * NE, NE,
                             BH, 1.f);
  launch_gemm<1, 0, 256, 64>(stream, attn, q_bf, nullptr, dk_bf, kN, kE, kN,
                             kN, kE, kE, kH, (long long)kH * NN2, NN2,
                             (long long)kH * NE, NE, (long long)kH * NE, NE,
                             BH, 1.f);

  // dwq = t_qk^T @ dq ; dwk = t_qk^T @ dk ; dwv = t_v^T @ dv
  launch_gemm<1, 0, 256, 64>(stream, tqk_bf, dq_bf, dwq_o, nullptr, kD, kE,
                             kN, kD, kE, kE, kH, ND, 0, (long long)kH * NE,
                             NE, (long long)kH * DE, DE, BH, 1.f);
  launch_gemm<1, 0, 256, 64>(stream, tqk_bf, dk_bf, dwk_o, nullptr, kD, kE,
                             kN, kD, kE, kE, kH, ND, 0, (long long)kH * NE,
                             NE, (long long)kH * DE, DE, BH, 1.f);
  launch_gemm<1, 0, 256, 64>(stream, tv_bf, dv_bf, dwv_o, nullptr, kD, kE,
                             kN, kD, kE, kE, kH, ND, 0, (long long)kH * NE,
                             NE, (long long)kH * DE, DE, BH, 1.f);

  // dwo = o_wo^T @ error  (64 x D per b,h)
  launch_gemm<1, 0, 128, 128>(stream, o_wo, err_bf, dwo_o, nullptr, kE, kD,
                              kN, kE, kD, kD, kH, (long long)kH * NE, NE, ND,
                              0, (long long)kH * DE, DE, BH, 1.f);

  // Newton-Schulz(5) on dwv and dwo, in place in d_out
  ns5_kernel<<<dim3(64), 256, 0, stream>>>(dwv_o, dwo_o, X0g, X1g);
}
